// LearnableTensorProduct_1030792151278
// MI455X (gfx1250) — compile-verified, matrix operand order checked by reading, not run on hardware
//
#include <hip/hip_runtime.h>
#include <hip/hip_bf16.h>

typedef __attribute__((ext_vector_type(16))) _Float16 v16h;
typedef __attribute__((ext_vector_type(8)))  float    v8f;

#define ZROWS 2048
#define DIMF  144
#define IPAD  160     // i (GEMM K-dim) padded to 5*32
#define NJT   9       // 144/16 j tiles
#define NKT   5       // 160/32 k-steps
#define NJOBS (144*128)

// ---- path tables (reference order: lo in {0,1,2}, (l1,l2) loop order) ----
__constant__ int P_LO[15]   = {0,0,0, 1,1,1,1,1,1, 2,2,2,2,2,2};
__constant__ int P_L1[15]   = {0,1,2, 0,1,1,1,2,2, 0,1,1,2,2,2};
__constant__ int P_L2[15]   = {0,1,2, 1,0,1,2,1,2, 2,1,2,0,1,2};
__constant__ int P_INLO[15] = {0,1,2, 0,1,2,3,4,5, 0,1,2,3,4,5};
__constant__ int C_PSZ[15]  = {1,9,25, 9,9,27,45,45,75, 25,45,75,25,75,125};   // d1*d2*d3
__constant__ int C_CGOFF[15]= {0,1,10, 35,44,53,80,125,170, 245,270,315,390,415,490}; // 615 total

// ================= CG construction (runs once, 15 threads) =================
__device__ double factd(int n){ double r = 1.0; for (int i = 2; i <= n; ++i) r *= (double)i; return r; }

__device__ double cg_cplx(int l1,int m1,int l2,int m2,int l3,int m3){
  if (m1 + m2 != m3) return 0.0;
  double pref = ((2.0*l3+1.0)*factd(l3+l1-l2)*factd(l3-l1+l2)*factd(l1+l2-l3))/factd(l1+l2+l3+1);
  pref = sqrt(pref);
  pref *= sqrt(factd(l3+m3)*factd(l3-m3)*factd(l1-m1)*factd(l1+m1)*factd(l2-m2)*factd(l2+m2));
  double s = 0.0;
  for (int k = 0; k <= 8; ++k) {
    int a = l1+l2-l3-k, b = l1-m1-k, c = l2+m2-k, d = l3-l2+m1+k, e = l3-l1-m2+k;
    if (a<0 || b<0 || c<0 || d<0 || e<0) continue;
    double term = 1.0/(factd(k)*factd(a)*factd(b)*factd(c)*factd(d)*factd(e));
    s += (k & 1) ? -term : term;
  }
  return pref * s;
}

// Q[a, mi] complex->real basis change (rows: real idx a, cols: complex idx mi=m+l)
__device__ void qel(int l, int a, int mi, double* qr, double* qi){
  *qr = 0.0; *qi = 0.0;
  const double s = 0.70710678118654752440;
  int ma = a - l, mc = mi - l;
  if (ma == 0) { if (mc == 0) *qr = 1.0; }
  else if (ma > 0) {
    int m = ma; double sg = (m & 1) ? -1.0 : 1.0;
    if (mc == m)       *qr = sg * s;
    else if (mc == -m) *qr = s;
  } else {
    int m = -ma; double sg = (m & 1) ? -1.0 : 1.0;
    if (mc == m)       *qi = -sg * s;
    else if (mc == -m) *qi = s;
  }
}

__global__ void cg_build(float* __restrict__ cg){
  int p = threadIdx.x;
  if (p >= 15) return;
  int lo = P_LO[p], l1 = P_L1[p], l2 = P_L2[p];
  int d1 = 2*l1+1, d2 = 2*l2+1, d3 = 2*lo+1;
  int n = d1*d2*d3;
  double re[125], im[125];
  for (int a = 0; a < d1; ++a)
    for (int b = 0; b < d2; ++b)
      for (int c = 0; c < d3; ++c) {
        double sr = 0.0, si = 0.0;
        for (int m1 = -l1; m1 <= l1; ++m1)
          for (int m2 = -l2; m2 <= l2; ++m2) {
            int m3 = m1 + m2;
            if (m3 < -lo || m3 > lo) continue;
            double t = cg_cplx(l1, m1, l2, m2, lo, m3);
            if (t == 0.0) continue;
            double q1r,q1i,q2r,q2i,q3r,q3i;
            qel(l1, a, m1+l1, &q1r, &q1i);
            qel(l2, b, m2+l2, &q2r, &q2i);
            qel(lo, c, m3+lo, &q3r, &q3i);
            double pr = q1r*q2r - q1i*q2i;
            double pi = q1r*q2i + q1i*q2r;
            double rr = pr*q3r + pi*q3i;     // (pr + i*pi) * conj(q3)
            double ri = pi*q3r - pr*q3i;
            sr += t * rr; si += t * ri;
          }
        int idx = (a*d2 + b)*d3 + c;
        re[idx] = sr; im[idx] = si;
      }
  double nr = 0.0, ni = 0.0;
  for (int i = 0; i < n; ++i) { nr += re[i]*re[i]; ni += im[i]*im[i]; }
  double* sel = (nr >= ni) ? re : im;
  double nrm  = sqrt((nr >= ni) ? nr : ni);
  int am = 0; double best = -1.0;
  for (int i = 0; i < n; ++i) { double v = fabs(sel[i]); if (v > best) { best = v; am = i; } }
  double sgn = (sel[am] >= 0.0) ? 1.0 : -1.0;
  float* dst = cg + C_CGOFF[p];
  for (int i = 0; i < n; ++i) dst[i] = (float)(sel[i] * sgn / nrm);
}

// ================= helpers =================
__global__ void zero_h(_Float16* __restrict__ p, int n){
  int t = blockIdx.x*blockDim.x + threadIdx.x;
  if (t < n) p[t] = (_Float16)0.0f;
}

__global__ void cvt_f1(const float* __restrict__ f, _Float16* __restrict__ o){
  int t = blockIdx.x*blockDim.x + threadIdx.x;
  if (t >= ZROWS*IPAD) return;
  int z = t / IPAD, col = t % IPAD;
  o[t] = (col < DIMF) ? (_Float16)f[z*DIMF + col] : (_Float16)0.0f;
}

// Scatter scale*W*C into WMMA-B-packed f16 buffer: [k][jt][kt][32 lanes][16 halves]
__global__ void fill_b(const float* __restrict__ w0, const float* __restrict__ w1,
                       const float* __restrict__ w2, const float* __restrict__ cg,
                       _Float16* __restrict__ Bp){
  int tid = blockIdx.x*blockDim.x + threadIdx.x;
  int p = 0, acc = 0;
  for (p = 0; p < 15; ++p) { int cnt = C_PSZ[p] * 4096; if (tid < acc + cnt) break; acc += cnt; }
  if (p >= 15) return;
  int e = tid - acc;
  int lo = P_LO[p], l1 = P_L1[p], l2 = P_L2[p];
  int d1 = 2*l1+1, d2 = 2*l2+1, d3 = 2*lo+1;
  int c = e % d3; e /= d3;
  int b = e % d2; e /= d2;
  int a = e % d1; e /= d1;
  int v = e & 15; e >>= 4;
  int u = e & 15; e >>= 4;
  int w = e;                                   // 0..15
  const float* W; int rowlen;
  if      (lo == 0) { W = w0; rowlen = 3*256; }
  else if (lo == 1) { W = w1; rowlen = 6*256; }
  else              { W = w2; rowlen = 6*256; }
  float wval = W[w*rowlen + P_INLO[p]*256 + u*16 + v];
  float cgv  = cg[C_CGOFF[p] + (a*d2 + b)*d3 + c];
  int   np   = (lo == 0) ? 3 : 6;
  float scale = sqrtf((float)d3 / ((float)np * 256.0f));
  float val = scale * wval * cgv;
  const int OFFl[3] = {0, 16, 64};
  int i  = OFFl[l1] + u*d1 + a;                // GEMM K index (0..143)
  int j  = OFFl[l2] + v*d2 + b;                // GEMM N index
  int kk = OFFl[lo] + w*d3 + c;                // output channel
  int kt = i >> 5, r = i & 31, jt = j >> 4, col = j & 15;
  // 16-bit B (32x16) VGPR layout: lanes 0-15 -> rows {0-7,16-23}, lanes 16-31 -> rows {8-15,24-31}
  int L, s;
  if      (r <  8) { L = col;      s = r;      }
  else if (r < 16) { L = col + 16; s = r - 8;  }
  else if (r < 24) { L = col;      s = r - 8;  }
  else             { L = col + 16; s = r - 16; }
  size_t idx = (((size_t)kk*NJT + jt)*NKT + kt)*512 + (size_t)L*16 + s;
  Bp[idx] = (_Float16)val;
}

// ================= main GEMM: one wave per (k, z-tile) =================
__global__ __launch_bounds__(256) void tp_main(const _Float16* __restrict__ F1h,
                                               const float*    __restrict__ F2,
                                               const _Float16* __restrict__ Bp,
                                               float*          __restrict__ out){
  int lane  = threadIdx.x & 31;
  int wave  = threadIdx.x >> 5;
  int job   = blockIdx.x*8 + wave;     // 0..18431
  int ztile = job / 144;
  int k     = job % 144;
  int zbase = ztile * 16;
  int half  = lane >> 4;               // 0/1
  int lcol  = lane & 15;

  // A operand (16-bit, 16x32): lane<16 gets K {0-7,16-23}, lane>=16 gets K {8-15,24-31}
  v16h A[NKT];
  const _Float16* f1row = F1h + (size_t)(zbase + lcol) * IPAD;
  #pragma unroll
  for (int kt = 0; kt < NKT; ++kt) {
    const _Float16* p0 = f1row + kt*32 + (half ? 8 : 0);
    #pragma unroll
    for (int t = 0; t < 8; ++t) A[kt][t]     = p0[t];
    #pragma unroll
    for (int t = 0; t < 8; ++t) A[kt][8 + t] = p0[16 + t];
  }

  // Per-lane partial sums; cross-lane reduction hoisted to the very end.
  float pp[8] = {0.f,0.f,0.f,0.f,0.f,0.f,0.f,0.f};
  const _Float16* bk  = Bp + (size_t)k * (NJT*NKT*512);
  const float*    f2p = F2 + (size_t)(zbase + half*8)*DIMF + lcol;

  for (int jt = 0; jt < NJT; ++jt) {
    v8f c = {};
    const _Float16* bbase = bk + (size_t)jt*(NKT*512) + (size_t)lane*16;
    #pragma unroll
    for (int kt = 0; kt < NKT; ++kt) {
      v16h b = *(const v16h*)(bbase + kt*512);
      c = __builtin_amdgcn_wmma_f32_16x16x32_f16(false, A[kt], false, b,
                                                 (short)0, c, false, false);
    }
    // C layout: VGPR r, half h -> z-row r+8h; lane%16 -> j.
    // Each jt hits a disjoint j range per lane, so just FMAC per-lane partials.
    #pragma unroll
    for (int r = 0; r < 8; ++r)
      pp[r] += c[r] * f2p[(size_t)r*DIMF + jt*16];
  }

  // One butterfly per job: sum over the 16 lanes of each half (xor bits 0..3).
  #pragma unroll
  for (int r = 0; r < 8; ++r) {
    float s = pp[r];
    s += __shfl_xor(s, 1, 32);
    s += __shfl_xor(s, 2, 32);
    s += __shfl_xor(s, 4, 32);
    s += __shfl_xor(s, 8, 32);
    pp[r] = s;
  }
  if (lcol == 0) {
    #pragma unroll
    for (int r = 0; r < 8; ++r)
      out[(size_t)(zbase + r + half*8)*DIMF + k] = pp[r];
  }
}

extern "C" void kernel_launch(void* const* d_in, const int* in_sizes, int n_in,
                              void* d_out, int out_size, void* d_ws, size_t ws_size,
                              hipStream_t stream) {
  const float* f1 = (const float*)d_in[0];
  const float* f2 = (const float*)d_in[1];
  const float* w0 = (const float*)d_in[2];
  const float* w1 = (const float*)d_in[3];
  const float* w2 = (const float*)d_in[4];
  float* out = (float*)d_out;

  char* ws = (char*)d_ws;
  float*     cg  = (float*)ws;                               // 615 floats (4 KB reserved)
  _Float16*  F1h = (_Float16*)(ws + 4096);                   // 2048*160 halfs = 640 KB
  _Float16*  Bp  = (_Float16*)(ws + 4096 + ZROWS*IPAD*2);    // 144*9*5*512 halfs = 6.33 MB

  cg_build<<<1, 32, 0, stream>>>(cg);

  const int nB = 144 * NJT * NKT * 512;
  zero_h<<<(nB + 255)/256, 256, 0, stream>>>(Bp, nB);

  cvt_f1<<<(ZROWS*IPAD + 255)/256, 256, 0, stream>>>(f1, F1h);

  const int nFill = 615 * 4096;   // sum over paths of 16^3 * d1*d2*d3
  fill_b<<<(nFill + 255)/256, 256, 0, stream>>>(w0, w1, w2, cg, Bp);

  tp_main<<<NJOBS/8, 256, 0, stream>>>(F1h, f2, Bp, out);
}
